// ScaledDotProductAttention_84902913507840
// MI455X (gfx1250) — matrix operand order repaired and checked by reading, not compiled, not run on hardware
//
#include <hip/hip_runtime.h>

typedef __attribute__((ext_vector_type(16))) _Float16 v16h;
typedef __attribute__((ext_vector_type(8)))  _Float16 v8h;
typedef __attribute__((ext_vector_type(4)))  _Float16 v4h;
typedef __attribute__((ext_vector_type(8)))  float    v8f;
typedef __attribute__((ext_vector_type(4)))  float    v4f;

#define NB      4
#define NH      16
#define S_LEN   2048
#define D_DIM   128
#define BLOCK_Q 128     // 8 waves * 16 query rows
#define KT      32      // key tile
#define WAVES   8
#define NEG_BIG (-1.0e30f)

__device__ __forceinline__ v8f wmma_f16(const v16h& a, const v16h& b, const v8f& c) {
    return __builtin_amdgcn_wmma_f32_16x16x32_f16(false, a, false, b, (short)0, c, false, false);
}

__global__ __launch_bounds__(256)
void sdpa_fwd_kernel(const float* __restrict__ Q, const float* __restrict__ K,
                     const float* __restrict__ V, float* __restrict__ Ctx,
                     float* __restrict__ W)
{
    __shared__ _Float16 sK[KT * D_DIM];       // [key][d]   8KB
    __shared__ _Float16 sVt[D_DIM * KT];      // [d][key]   8KB
    __shared__ _Float16 sP[WAVES][16 * KT];   // per-wave P transpose bounce, 8KB

    const int tid  = threadIdx.x;
    const int wave = tid >> 5;
    const int lane = tid & 31;
    const int g    = lane >> 4;   // half-wave group
    const int ln   = lane & 15;

    const int bh   = blockIdx.x / (S_LEN / BLOCK_Q);
    const int qblk = blockIdx.x % (S_LEN / BLOCK_Q);
    const int m0b  = qblk * BLOCK_Q;
    const int m0   = m0b + wave * 16;     // this wave's first query row
    const int myLast = m0 + 15;

    const size_t baseQKV = (size_t)bh * S_LEN * D_DIM;
    const float* Qp = Q + baseQKV;
    const float* Kp = K + baseQKV;
    const float* Vp = V + baseQKV;
    float*       Cp = Ctx + baseQKV;
    float*       Wp = W + (size_t)bh * S_LEN * S_LEN;

    // ---- load my Q strip (16 x 128) into WMMA A-layout registers ----
    // A 16x32 f16 layout: lane L -> row M = L%16; group g=L/16:
    //   elems 0..7  -> K = 32c + 8g + (0..7)
    //   elems 8..15 -> K = 32c + 16 + 8g + (0..7)
    // Q is read exactly once per block -> non-temporal loads (don't pollute L2,
    // which we want to keep holding the K/V tiles shared by 16 query-blocks).
    v16h qa[4];
    {
        const float* qrow = Qp + (size_t)(m0 + ln) * D_DIM;
        #pragma unroll
        for (int c = 0; c < 4; ++c) {
            v4f f0 = __builtin_nontemporal_load((const v4f*)(qrow + c * 32 + g * 8 + 0));
            v4f f1 = __builtin_nontemporal_load((const v4f*)(qrow + c * 32 + g * 8 + 4));
            v4f f2 = __builtin_nontemporal_load((const v4f*)(qrow + c * 32 + 16 + g * 8 + 0));
            v4f f3 = __builtin_nontemporal_load((const v4f*)(qrow + c * 32 + 16 + g * 8 + 4));
            v16h a;
            #pragma unroll
            for (int j = 0; j < 4; ++j) {
                a[j]      = (_Float16)f0[j];
                a[4 + j]  = (_Float16)f1[j];
                a[8 + j]  = (_Float16)f2[j];
                a[12 + j] = (_Float16)f3[j];
            }
            qa[c] = a;
        }
    }

    const float scale = 0.08838834764831845f;   // 1/sqrt(128)
    const int nkt = (m0b + BLOCK_Q) / KT;       // causal tile count for this block

    float mrun[8], lrun[8];
    #pragma unroll
    for (int r = 0; r < 8; ++r) { mrun[r] = NEG_BIG; lrun[r] = 0.0f; }

    // ================= pass 1: softmax stats =================
    for (int kt = 0; kt < nkt; ++kt) {
        const int kt0 = kt * KT;

        // cooperative stage: K tile (32 keys x 128 d) fp32 -> fp16 LDS
        #pragma unroll
        for (int t = 0; t < 4; ++t) {
            int idx = tid + t * 256;           // vec4 id, 0..1023
            int row = idx >> 5;                // 32 vec4 per row
            int c4  = idx & 31;
            v4f f = *(const v4f*)(Kp + (size_t)(kt0 + row) * D_DIM + c4 * 4);
            v4h h = { (_Float16)f[0], (_Float16)f[1], (_Float16)f[2], (_Float16)f[3] };
            *(v4h*)&sK[row * D_DIM + c4 * 4] = h;
        }
        __syncthreads();

        if (kt0 <= myLast) {   // wave-uniform => EXEC all-1 for WMMA
            #pragma unroll
            for (int s = 0; s < 2; ++s) {
                v8f c8 = {};
                #pragma unroll
                for (int c = 0; c < 4; ++c) {
                    // B 32x16 layout: lane L -> col N = L%16; k = 8g + (0..7), 16 + 8g + (0..7)
                    const v8h kl = *(const v8h*)&sK[(s * 16 + ln) * D_DIM + c * 32 + g * 8];
                    const v8h kh = *(const v8h*)&sK[(s * 16 + ln) * D_DIM + c * 32 + 16 + g * 8];
                    const v16h kb = __builtin_shufflevector(kl, kh, 0,1,2,3,4,5,6,7,8,9,10,11,12,13,14,15);
                    c8 = wmma_f16(qa[c], kb, c8);
                }
                const int keyn = kt0 + s * 16 + ln;
                #pragma unroll
                for (int r = 0; r < 8; ++r) {
                    const int mrow = m0 + r + 8 * g;
                    float sc = c8[r] * scale;
                    c8[r] = (keyn <= mrow) ? sc : NEG_BIG;
                }
                #pragma unroll
                for (int r = 0; r < 8; ++r) {
                    float x = c8[r];
                    #pragma unroll
                    for (int off = 1; off < 16; off <<= 1)
                        x = fmaxf(x, __shfl_xor(x, off, 32));
                    float newm = fmaxf(mrun[r], x);
                    float e = __expf(c8[r] - newm);
                    #pragma unroll
                    for (int off = 1; off < 16; off <<= 1)
                        e += __shfl_xor(e, off, 32);
                    lrun[r] = lrun[r] * __expf(mrun[r] - newm) + e;
                    mrun[r] = newm;
                }
            }
        }
        __syncthreads();
    }

    float linv[8];
    #pragma unroll
    for (int r = 0; r < 8; ++r) linv[r] = (lrun[r] > 0.0f) ? (1.0f / lrun[r]) : 0.0f;

    v8f acc[8];
    #pragma unroll
    for (int nt = 0; nt < 8; ++nt) acc[nt] = (v8f){};

    // ================= pass 2: weights + context =================
    for (int kt = 0; kt < nkt; ++kt) {
        const int kt0 = kt * KT;

        // stage K (row-major f16) and V (transposed [d][key] f16)
        #pragma unroll
        for (int t = 0; t < 4; ++t) {
            int idx = tid + t * 256;
            int row = idx >> 5;
            int c4  = idx & 31;
            v4f fk = *(const v4f*)(Kp + (size_t)(kt0 + row) * D_DIM + c4 * 4);
            v4h hk = { (_Float16)fk[0], (_Float16)fk[1], (_Float16)fk[2], (_Float16)fk[3] };
            *(v4h*)&sK[row * D_DIM + c4 * 4] = hk;
            v4f fv = *(const v4f*)(Vp + (size_t)(kt0 + row) * D_DIM + c4 * 4);
            #pragma unroll
            for (int j = 0; j < 4; ++j)
                sVt[(c4 * 4 + j) * KT + row] = (_Float16)fv[j];
        }
        __syncthreads();

        if (kt0 <= myLast) {
            #pragma unroll
            for (int s = 0; s < 2; ++s) {
                v8f c8 = {};
                #pragma unroll
                for (int c = 0; c < 4; ++c) {
                    const v8h kl = *(const v8h*)&sK[(s * 16 + ln) * D_DIM + c * 32 + g * 8];
                    const v8h kh = *(const v8h*)&sK[(s * 16 + ln) * D_DIM + c * 32 + 16 + g * 8];
                    const v16h kb = __builtin_shufflevector(kl, kh, 0,1,2,3,4,5,6,7,8,9,10,11,12,13,14,15);
                    c8 = wmma_f16(qa[c], kb, c8);
                }
                const int keyn = kt0 + s * 16 + ln;
                #pragma unroll
                for (int r = 0; r < 8; ++r) {
                    const int mrow = m0 + r + 8 * g;
                    float sc = c8[r] * scale;
                    sc = (keyn <= mrow) ? sc : NEG_BIG;
                    float p = __expf(sc - mrun[r]) * linv[r];
                    // streaming, write-once output: bypass-friendly NT store
                    __builtin_nontemporal_store(p, &Wp[(size_t)mrow * S_LEN + keyn]);
                    sP[wave][(r + 8 * g) * KT + s * 16 + ln] = (_Float16)p;
                }
            }
            // read P back in A-layout (16 x 32, K = keys)
            v8h lo = *(const v8h*)&sP[wave][ln * KT + g * 8];
            v8h hi = *(const v8h*)&sP[wave][ln * KT + 16 + g * 8];
            v16h pa;
            #pragma unroll
            for (int j = 0; j < 8; ++j) { pa[j] = lo[j]; pa[8 + j] = hi[j]; }
            #pragma unroll
            for (int nt = 0; nt < 8; ++nt) {
                // B 32x16 (keys x d): lane L -> d col N = L%16, key k = 8g + (0..7), 16 + 8g + (0..7)
                const v8h vl = *(const v8h*)&sVt[(nt * 16 + ln) * KT + g * 8];
                const v8h vh = *(const v8h*)&sVt[(nt * 16 + ln) * KT + 16 + g * 8];
                const v16h vb = __builtin_shufflevector(vl, vh, 0,1,2,3,4,5,6,7,8,9,10,11,12,13,14,15);
                acc[nt] = wmma_f16(pa, vb, acc[nt]);
            }
        } else {
            // tile fully masked for this wave's rows: weights are exactly 0
            #pragma unroll
            for (int s = 0; s < 2; ++s)
                #pragma unroll
                for (int r = 0; r < 8; ++r)
                    __builtin_nontemporal_store(
                        0.0f, &Wp[(size_t)(m0 + r + 8 * g) * S_LEN + kt0 + s * 16 + ln]);
        }
        __syncthreads();
    }

    // write context (C/D layout: lane L -> d = 16nt + L%16, row = r + 8g)
    #pragma unroll
    for (int nt = 0; nt < 8; ++nt)
        #pragma unroll
        for (int r = 0; r < 8; ++r)
            __builtin_nontemporal_store(
                acc[nt][r], &Cp[(size_t)(m0 + r + 8 * g) * D_DIM + nt * 16 + ln]);

    // zero-fill the untouched upper-triangle tail of the weight matrix
    const int tail0 = m0b + BLOCK_Q;
    const int tailN = S_LEN - tail0;           // columns never visited by this block
    if (tailN > 0) {
        const int vec4PerRow = tailN >> 2;
        const int total = BLOCK_Q * vec4PerRow;
        const v4f z4 = {};
        for (int i = tid; i < total; i += 256) {
            int row = i / vec4PerRow;
            int c4  = i % vec4PerRow;
            __builtin_nontemporal_store(
                z4, (v4f*)&Wp[(size_t)(m0b + row) * S_LEN + tail0 + c4 * 4]);
        }
    }
}

extern "C" void kernel_launch(void* const* d_in, const int* in_sizes, int n_in,
                              void* d_out, int out_size, void* d_ws, size_t ws_size,
                              hipStream_t stream) {
    (void)in_sizes; (void)n_in; (void)out_size; (void)d_ws; (void)ws_size;
    const float* q = (const float*)d_in[0];
    const float* k = (const float*)d_in[1];
    const float* v = (const float*)d_in[2];
    // d_in[3] is the boolean causal mask; causality is computed analytically.
    float* ctx   = (float*)d_out;
    float* attnW = ctx + (size_t)NB * NH * S_LEN * D_DIM;

    dim3 grid(NB * NH * (S_LEN / BLOCK_Q));
    dim3 block(256);
    sdpa_fwd_kernel<<<grid, block, 0, stream>>>(q, k, v, ctx, attnW);
}
